// WideResNet_SDN_8117488189563
// MI455X (gfx1250) — matrix operand order repaired and checked by reading, not compiled, not run on hardware
//
#include <hip/hip_runtime.h>
#include <hip/hip_bf16.h>

// ---------------------------------------------------------------------------
// Types for WMMA bf16 path (gfx1250 / CDNA5, wave32)
// ---------------------------------------------------------------------------
typedef __attribute__((ext_vector_type(16))) __bf16 bf16x16;
typedef __attribute__((ext_vector_type(8)))  float  f32x8;
typedef __attribute__((ext_vector_type(8)))  unsigned short u16x8;
typedef __attribute__((ext_vector_type(4)))  int    i32x4;

#if defined(__gfx1250__) && __has_builtin(__builtin_amdgcn_global_load_async_to_lds_b128) && __has_builtin(__builtin_amdgcn_s_wait_asynccnt)
#define USE_ASYNC_LDS 1
#else
#define USE_ASYNC_LDS 0
#endif

__device__ __forceinline__ unsigned short f32_to_bf16_rne(float f) {
    unsigned int u = __float_as_uint(f);
    unsigned int r = u + 0x7FFFu + ((u >> 16) & 1u);   // round-to-nearest-even
    return (unsigned short)(r >> 16);
}

// ---------------------------------------------------------------------------
// BN prep: scale = g/sqrt(v+eps), shift = b - m*scale
// ---------------------------------------------------------------------------
__global__ void bn_prep_kernel(const float* __restrict__ g, const float* __restrict__ b,
                               const float* __restrict__ m, const float* __restrict__ v,
                               float* __restrict__ scale, float* __restrict__ shift, int C) {
    int c = blockIdx.x * blockDim.x + threadIdx.x;
    if (c < C) {
        float s = g[c] * rsqrtf(v[c] + 1e-5f);
        scale[c] = s;
        shift[c] = b[c] - m[c] * s;
    }
}

// ---------------------------------------------------------------------------
// Weight pack: OIHW f32 -> bf16 fragment-order [chunk][Co][32],
//   element (kpad, co) stored at ((kpad>>5)*Co + co)*32 + (kpad&31),
//   K index = (ky*ks+kx)*Ci + ci, zero-padded to Kpad = ceil(K/32)*32.
// For a 64-wide Co tile, one K-chunk of B is a contiguous 4KB block in the
// exact [n][k] layout the WMMA B-fragment loads want.
// ---------------------------------------------------------------------------
__global__ void pack_w_kernel(const float* __restrict__ w, unsigned short* __restrict__ wp,
                              int Co, int Ci, int ks, int Kpad) {
    long total = (long)Kpad * Co;
    long i = blockIdx.x * (long)blockDim.x + threadIdx.x;
    if (i >= total) return;
    int co   = (int)(i % Co);
    int kpad = (int)(i / Co);
    int K = ks * ks * Ci;
    unsigned short val = 0;
    if (kpad < K) {
        int ci   = kpad % Ci;
        int cell = kpad / Ci;
        int kx = cell % ks, ky = cell / ks;
        val = f32_to_bf16_rne(w[((long)(co * Ci + ci) * ks + ky) * ks + kx]);
    }
    wp[((long)(kpad >> 5) * Co + co) * 32 + (kpad & 31)] = val;
}

// ---------------------------------------------------------------------------
// Implicit-GEMM conv via v_wmma_f32_16x16x32_bf16.
//   GEMM: M = N*Ho*Wo output pixels, Ndim = Co, K = KS*KS*Ci
//   Tile: 128(M) x 64(N), 8 waves; each wave: 1 M-tile x 4 N-subtiles.
//   Double-buffered LDS; B staged via async global->LDS when available.
//   All divides removed: Ci/Wo/HW are powers of two (shift/mask params).
// ---------------------------------------------------------------------------
#define CONV_BM 128
#define CONV_BN 64
#define CONV_BK 32

template <int KS, int BN>
__launch_bounds__(256)
__global__ void conv_wmma_kernel(
    const float* __restrict__ x, const unsigned short* __restrict__ wp,
    const float* __restrict__ bias,
    const float* __restrict__ bnScale, const float* __restrict__ bnShift,
    float* __restrict__ y,
    int Hin, int Win, int Ci, int ciShift, int Co,
    int Ho, int Wo, int woShift, int hwShift,
    int stride, int accum)
{
    __shared__ __align__(16) unsigned short As[2][CONV_BM * CONV_BK]; // [m][k]
    __shared__ __align__(16) unsigned short Bs[2][CONV_BN * CONV_BK]; // [n][k]

    const int K    = KS * KS * Ci;
    const int tid  = threadIdx.x;
    const int lane = tid & 31;
    const int wave = tid >> 5;
    const int mBase = blockIdx.x * CONV_BM;
    const int nBase = blockIdx.y * CONV_BN;
    const int ciMask = Ci - 1;
    const int woMask = Wo - 1;
    const int hwMask = (1 << hwShift) - 1;
    const int pad = (KS == 3) ? 1 : 0;

    // ---- per-thread A-gather coordinates (fixed across K loop)
    const int a_m    = tid >> 1;                 // local row 0..127
    const int a_kOff = (tid & 1) * 16;           // K sub-offset 0/16
    const int a_mg   = mBase + a_m;
    const int a_n    = a_mg >> hwShift;
    const int a_rem  = a_mg & hwMask;
    const int a_ho   = a_rem >> woShift;
    const int a_wo   = a_rem & woMask;
    const int hBase  = a_ho * stride - pad;
    const int wBase  = a_wo * stride - pad;
    const float* __restrict__ xn = x + (long)a_n * Hin * Win * Ci;

    // stage A chunk (im2col + BN + ReLU + bf16) into buffer `buf`
    auto stageA = [&](int k0, int buf) {
        unsigned short* dst = &As[buf][a_m * CONV_BK + a_kOff];
#pragma unroll
        for (int j = 0; j < 16; ++j) {
            int kk = k0 + a_kOff + j;
            float val = 0.0f;
            if (kk < K) {
                int ci   = kk & ciMask;
                int cell = kk >> ciShift;          // 0..KS*KS-1
                int ky = (KS == 3) ? (cell / 3) : 0;
                int kx = (KS == 3) ? (cell - ky * 3) : 0;
                int h = hBase + ky, w = wBase + kx;
                if (h >= 0 && h < Hin && w >= 0 && w < Win) {
                    val = xn[((long)h * Win + w) * Ci + ci];
                    if (BN) val = val * bnScale[ci] + bnShift[ci];
                    val = fmaxf(val, 0.0f);
                }
            }
            dst[j] = f32_to_bf16_rne(val);
        }
    };

    // stage B chunk: contiguous 4KB copy from fragment-ordered weights
    auto stageB = [&](int k0, int buf) {
        const unsigned short* src = wp + ((long)(k0 >> 5) * Co + nBase) * 32;
#if USE_ASYNC_LDS
        __builtin_amdgcn_global_load_async_to_lds_b128(
            (__attribute__((address_space(1))) i32x4*)(src + tid * 8),
            (__attribute__((address_space(3))) i32x4*)(&Bs[buf][tid * 8]), 0, 0);
#else
        ((u16x8*)&Bs[buf][0])[tid] = ((const u16x8*)src)[tid];
#endif
    };

    f32x8 acc[4];
#pragma unroll
    for (int s = 0; s < 4; ++s)
#pragma unroll
        for (int r = 0; r < 8; ++r) acc[s][r] = 0.0f;

    const int nChunks = (K + CONV_BK - 1) >> 5;
    stageA(0, 0);
    stageB(0, 0);

    const int a_row = wave * 16 + (lane & 15);
    const int a_kb  = (lane < 16) ? 0 : 8;
    const int b_kb  = (lane < 16) ? 0 : 8;
    const int ncol  = lane & 15;

    for (int c = 0; c < nChunks; ++c) {
        const int buf = c & 1;
#if USE_ASYNC_LDS
        __builtin_amdgcn_s_wait_asynccnt(0);
#endif
        __syncthreads();
        if (c + 1 < nChunks) {
            stageA((c + 1) * CONV_BK, buf ^ 1);
            stageB((c + 1) * CONV_BK, buf ^ 1);
        }
        // A 16x32 bf16 fragment: halves 0..7 = K[kb..kb+7], 8..15 = K[kb+16..kb+23]
        union { u16x8 h[2]; bf16x16 v; } afrag;
        const unsigned short* ap = &As[buf][a_row * CONV_BK + a_kb];
        afrag.h[0] = *(const u16x8*)(ap);
        afrag.h[1] = *(const u16x8*)(ap + 16);
        // B 32x16 bf16 fragment: halves 0..7 = K[kb..kb+7], 8..15 = K[kb+16..kb+23]
#pragma unroll
        for (int s = 0; s < 4; ++s) {
            union { u16x8 h[2]; bf16x16 v; } bfrag;
            const unsigned short* bp = &Bs[buf][(s * 16 + ncol) * CONV_BK + b_kb];
            bfrag.h[0] = *(const u16x8*)(bp);
            bfrag.h[1] = *(const u16x8*)(bp + 16);
            acc[s] = __builtin_amdgcn_wmma_f32_16x16x32_bf16(
                false, afrag.v, false, bfrag.v, (short)0, acc[s], false, false);
        }
    }

    // ---- epilogue: C/D layout (VGPR r: lanes 0-15 -> M=r, lanes 16-31 -> M=r+8)
#pragma unroll
    for (int r = 0; r < 8; ++r) {
        int mLoc = wave * 16 + r + ((lane < 16) ? 0 : 8);
        int m = mBase + mLoc;
        int n  = m >> hwShift;
        int rem = m & hwMask;
        int ho = rem >> woShift, wo = rem & woMask;
        long base = (((long)n * Ho + ho) * Wo + wo) * Co + nBase + ncol;
#pragma unroll
        for (int s = 0; s < 4; ++s) {
            long idx = base + s * 16;
            float val = acc[s][r] + bias[nBase + s * 16 + ncol];
            if (accum) val += y[idx];
            y[idx] = val;
        }
    }
}

// ---------------------------------------------------------------------------
// Init conv: NCHW(128,3,32,32) -> NHWC(128,32,32,16), 3x3 pad1 (tiny: scalar)
// ---------------------------------------------------------------------------
__global__ void init_conv_kernel(const float* __restrict__ x, const float* __restrict__ w,
                                 const float* __restrict__ b, float* __restrict__ y, int Nb) {
    int i = blockIdx.x * blockDim.x + threadIdx.x;
    int total = Nb * 32 * 32 * 16;
    if (i >= total) return;
    int co = i & 15; int t = i >> 4;
    int wo = t & 31; t >>= 5;
    int ho = t & 31; int n = t >> 5;
    float acc = b[co];
    for (int ci = 0; ci < 3; ++ci)
        for (int ky = 0; ky < 3; ++ky) {
            int h = ho - 1 + ky;
            if (h < 0 || h >= 32) continue;
            for (int kx = 0; kx < 3; ++kx) {
                int ww = wo - 1 + kx;
                if (ww < 0 || ww >= 32) continue;
                acc += x[((n * 3 + ci) * 32 + h) * 32 + ww] * w[((co * 3 + ci) * 3 + ky) * 3 + kx];
            }
        }
    y[i] = acc;
}

__global__ void add_kernel(float* __restrict__ y, const float* __restrict__ x, long n) {
    long i = blockIdx.x * (long)blockDim.x + threadIdx.x;
    if (i < n) y[i] += x[i];
}

// mixed pool: alpha*maxpool + (1-alpha)*avgpool, k x k -> 4x4, flatten c*16+py*4+px
__global__ void mixed_pool_kernel(const float* __restrict__ x, const float* __restrict__ alpha,
                                  float* __restrict__ pooled, int Nb, int S, int C, int k) {
    int i = blockIdx.x * blockDim.x + threadIdx.x;
    int total = Nb * C * 16;
    if (i >= total) return;
    int px = i & 3, py = (i >> 2) & 3;
    int t = i >> 4;
    int c = t % C, n = t / C;
    float mx = -3.402823466e38f, sum = 0.0f;
    for (int dy = 0; dy < k; ++dy)
        for (int dx = 0; dx < k; ++dx) {
            float v = x[(((long)n * S + py * k + dy) * S + px * k + dx) * C + c];
            mx = fmaxf(mx, v);
            sum += v;
        }
    float a = alpha[0];
    pooled[i] = a * mx + (1.0f - a) * sum / (float)(k * k);
}

__global__ void linear_kernel(const float* __restrict__ in, const float* __restrict__ w,
                              const float* __restrict__ b, float* __restrict__ out,
                              int Nb, int CI, int CO) {
    int i = blockIdx.x * blockDim.x + threadIdx.x;
    if (i >= Nb * CO) return;
    int j = i % CO, n = i / CO;
    float acc = b[j];
    for (int q = 0; q < CI; ++q) acc += in[(long)n * CI + q] * w[(long)j * CI + q];
    out[n * CO + j] = acc;
}

// final head pool: relu(bn(x)) then 8x8 avgpool -> vec[Nb,256]
__global__ void head_pool_kernel(const float* __restrict__ x, const float* __restrict__ scale,
                                 const float* __restrict__ shift, float* __restrict__ vec, int Nb) {
    int i = blockIdx.x * blockDim.x + threadIdx.x;
    if (i >= Nb * 256) return;
    int c = i & 255, n = i >> 8;
    float s = 0.0f;
    for (int p = 0; p < 64; ++p) {
        int hy = p >> 3, wx = p & 7;
        float v = x[(((long)n * 8 + hy) * 8 + wx) * 256 + c];
        s += fmaxf(v * scale[c] + shift[c], 0.0f);
    }
    vec[i] = s * (1.0f / 64.0f);
}

// ---------------------------------------------------------------------------
// Host orchestration
// ---------------------------------------------------------------------------
static inline int ilog2(int v) { int s = 0; while ((1 << s) < v) ++s; return s; }

extern "C" void kernel_launch(void* const* d_in, const int* in_sizes, int n_in,
                              void* d_out, int out_size, void* d_ws, size_t ws_size,
                              hipStream_t stream) {
    (void)in_sizes; (void)n_in; (void)out_size; (void)ws_size;
    auto F = [&](int i) { return (const float*)d_in[i]; };
    int idx = 0;
    const float* X    = F(idx++);
    const float* icvW = F(idx++);
    const float* icvB = F(idx++);

    struct Blk {
        int in_ch, ch, stride, size, has_sc, has_ic;
        const float *bn1[4], *w1, *b1, *bn2[4], *w2, *b2, *scw, *scb, *alpha, *lw, *lb;
    };
    Blk blk[15];
    const int stages[3][3] = {{64, 1, 32}, {128, 2, 16}, {256, 2, 8}};
    const int addm[3][5]   = {{0, 0, 1, 0, 1}, {0, 1, 0, 1, 0}, {1, 0, 1, 0, 0}};
    {
        int in_ch = 16, bi = 0;
        for (int s = 0; s < 3; ++s)
            for (int i2 = 0; i2 < 5; ++i2) {
                Blk& B = blk[bi++];
                B.ch = stages[s][0];
                B.stride = (i2 == 0) ? stages[s][1] : 1;
                B.size = stages[s][2];
                B.in_ch = in_ch;
                B.has_sc = (B.stride != 1 || B.in_ch != B.ch) ? 1 : 0;
                B.has_ic = addm[s][i2];
                in_ch = B.ch;
            }
    }
    for (int b = 0; b < 15; ++b) {
        Blk& B = blk[b];
        for (int j = 0; j < 4; ++j) B.bn1[j] = F(idx++);
        B.w1 = F(idx++); B.b1 = F(idx++);
        for (int j = 0; j < 4; ++j) B.bn2[j] = F(idx++);
        B.w2 = F(idx++); B.b2 = F(idx++);
        B.scw = B.scb = B.alpha = B.lw = B.lb = nullptr;
        if (B.has_sc) { B.scw = F(idx++); B.scb = F(idx++); }
        if (B.has_ic) { B.alpha = F(idx++); B.lw = F(idx++); B.lb = F(idx++); }
    }
    const float* ebn[4];
    for (int j = 0; j < 4; ++j) ebn[j] = F(idx++);
    const float* elW = F(idx++);
    const float* elB = F(idx++);

    // workspace carve (activations NHWC f32)
    char* ws = (char*)d_ws;
    auto carve = [&](size_t bytes) -> void* {
        void* p = ws;
        ws += (bytes + 255) & ~(size_t)255;
        return p;
    };
    float* actA = (float*)carve(128L * 32 * 32 * 64 * 4);   // 33.5 MB
    float* actB = (float*)carve(128L * 32 * 32 * 128 * 4);  // 67 MB (conv1 out at stage edge)
    float* actC = (float*)carve(128L * 32 * 32 * 64 * 4);   // 33.5 MB
    unsigned short* wpack = (unsigned short*)carve(2304L * 256 * 2);
    float* bnS = (float*)carve(256 * 4);
    float* bnH = (float*)carve(256 * 4);
    float* pooled = (float*)carve(128L * 4096 * 4);

    const int Nb = 128;
    float* out = (float*)d_out;

    {
        int total = Nb * 32 * 32 * 16;
        init_conv_kernel<<<(total + 255) / 256, 256, 0, stream>>>(X, icvW, icvB, actA, Nb);
    }

    int curH = 32, icCount = 0;
    for (int b = 0; b < 15; ++b) {
        Blk& B = blk[b];
        int H = curH, Ci = B.in_ch, Co = B.ch, st = B.stride;
        int Hout = H / st;
        int woS_in  = ilog2(H),   hwS_in  = 2 * woS_in;   // conv1 output = input spatial
        int woS_out = ilog2(Hout), hwS_out = 2 * woS_out;

        // conv1: relu(bn1(actA)) * W1 -> actB   (stride 1, pad 1)
        bn_prep_kernel<<<(Ci + 63) / 64, 64, 0, stream>>>(B.bn1[0], B.bn1[1], B.bn1[2], B.bn1[3], bnS, bnH, Ci);
        {
            int Kpad = (9 * Ci + 31) & ~31;
            long tot = (long)Kpad * Co;
            pack_w_kernel<<<(unsigned)((tot + 255) / 256), 256, 0, stream>>>(B.w1, wpack, Co, Ci, 3, Kpad);
            long M = (long)Nb * H * H;
            dim3 g((unsigned)(M / CONV_BM), (unsigned)(Co / CONV_BN));
            conv_wmma_kernel<3, 1><<<g, 256, 0, stream>>>(actA, wpack, B.b1, bnS, bnH, actB,
                H, H, Ci, ilog2(Ci), Co, H, H, woS_in, hwS_in, 1, 0);
        }
        // conv2: relu(bn2(actB)) * W2 -> actC   (stride st, pad 1)
        bn_prep_kernel<<<(Co + 63) / 64, 64, 0, stream>>>(B.bn2[0], B.bn2[1], B.bn2[2], B.bn2[3], bnS, bnH, Co);
        {
            int Kpad = (9 * Co + 31) & ~31;
            long tot = (long)Kpad * Co;
            pack_w_kernel<<<(unsigned)((tot + 255) / 256), 256, 0, stream>>>(B.w2, wpack, Co, Co, 3, Kpad);
            long M = (long)Nb * Hout * Hout;
            dim3 g((unsigned)(M / CONV_BM), (unsigned)(Co / CONV_BN));
            conv_wmma_kernel<3, 1><<<g, 256, 0, stream>>>(actB, wpack, B.b2, bnS, bnH, actC,
                H, H, Co, ilog2(Co), Co, Hout, Hout, woS_out, hwS_out, st, 0);
        }
        // shortcut
        if (B.has_sc) {
            int Kpad = (Ci + 31) & ~31;
            long tot = (long)Kpad * Co;
            pack_w_kernel<<<(unsigned)((tot + 255) / 256), 256, 0, stream>>>(B.scw, wpack, Co, Ci, 1, Kpad);
            long M = (long)Nb * Hout * Hout;
            dim3 g((unsigned)(M / CONV_BM), (unsigned)(Co / CONV_BN));
            conv_wmma_kernel<1, 0><<<g, 256, 0, stream>>>(actA, wpack, B.scb, nullptr, nullptr, actC,
                H, H, Ci, ilog2(Ci), Co, Hout, Hout, woS_out, hwS_out, st, 1);
        } else {
            long tot = (long)Nb * Hout * Hout * Co;
            add_kernel<<<(unsigned)((tot + 255) / 256), 256, 0, stream>>>(actC, actA, tot);
        }
        // rotate buffers: block output becomes next input
        float* t = actA; actA = actC; actC = t;
        curH = Hout;

        if (B.has_ic) {
            int S = B.size, k = S / 4, CI = Co * 16;
            int tot = Nb * CI;
            mixed_pool_kernel<<<(tot + 255) / 256, 256, 0, stream>>>(actA, B.alpha, pooled, Nb, S, Co, k);
            int tot2 = Nb * 10;
            linear_kernel<<<(tot2 + 255) / 256, 256, 0, stream>>>(pooled, B.lw, B.lb,
                                                                  out + icCount * Nb * 10, Nb, CI, 10);
            icCount++;
        }
    }

    // final head
    bn_prep_kernel<<<4, 64, 0, stream>>>(ebn[0], ebn[1], ebn[2], ebn[3], bnS, bnH, 256);
    {
        int tot = Nb * 256;
        head_pool_kernel<<<(tot + 255) / 256, 256, 0, stream>>>(actA, bnS, bnH, pooled, Nb);
        int tot2 = Nb * 10;
        linear_kernel<<<(tot2 + 255) / 256, 256, 0, stream>>>(pooled, elW, elB, out + 6 * Nb * 10, Nb, 256, 10);
    }
}
